// GraphSAGE_32117765439923
// MI455X (gfx1250) — matrix operand order repaired and checked by reading, not compiled, not run on hardware
//
#include <hip/hip_runtime.h>

#define NNODES 100000
#define NEDGES 1600000
#define DFEAT  128      // D_IN == D_HID == 128

typedef __attribute__((ext_vector_type(16))) _Float16 v16h;
typedef __attribute__((ext_vector_type(8)))  float    v8f;

// ---------------------------------------------------------------- utilities
__global__ void zero_f32_kernel(float* __restrict__ p, int n) {
    int i = blockIdx.x * blockDim.x + threadIdx.x;
    if (i < n) p[i] = 0.0f;
}

// Convert W [128][DOUT] f32 -> f16, pre-swizzled into this kernel's contiguous-k B-fragment
// layout: out[(((ct*4)+kc)*32 + lane)*16 + i] = W[kc*32 + 16*(lane>>4) + i][ct*16 + (lane&15)]
// so in the GEMM each lane's 16 f16 B values are one contiguous 32B load.
__global__ void swizzle_weight_kernel(const float* __restrict__ W,
                                      _Float16* __restrict__ out, int DOUT) {
    int idx = blockIdx.x * blockDim.x + threadIdx.x;
    if (idx >= 128 * DOUT) return;
    const int i    = idx & 15;
    const int lane = (idx >> 4) & 31;
    const int kc   = (idx >> 9) & 3;
    const int ct   = idx >> 11;
    const int k    = kc * 32 + 16 * (lane >> 4) + i;
    const int col  = ct * 16 + (lane & 15);
    out[idx] = (_Float16)W[(size_t)k * DOUT + col];
}

// deg[v] = number of edges with dst == v
__global__ void degree_kernel(const int* __restrict__ dst,
                              float* __restrict__ deg, int E) {
    int e = blockIdx.x * blockDim.x + threadIdx.x;
    if (e < E) atomicAdd(&deg[dst[e]], 1.0f);
}

// deg[i] <- 1 / max(deg[i], 1)
__global__ void invdeg_kernel(float* __restrict__ deg, int n) {
    int i = blockIdx.x * blockDim.x + threadIdx.x;
    if (i < n) deg[i] = 1.0f / fmaxf(deg[i], 1.0f);
}

// agg[dst[e]] += feat[src[e]]  — one edge per 32 lanes, float4 per lane
__global__ void scatter_add_kernel(const float* __restrict__ feat,
                                   const int* __restrict__ src,
                                   const int* __restrict__ dst,
                                   float* __restrict__ agg, int E) {
    int t = blockIdx.x * blockDim.x + threadIdx.x;
    int e = t >> 5;
    if (e >= E) return;
    int lane = t & 31;
    const float4 v = *(const float4*)(feat + (size_t)src[e] * DFEAT + lane * 4);
    float* a = agg + (size_t)dst[e] * DFEAT + lane * 4;
    atomicAdd(a + 0, v.x);
    atomicAdd(a + 1, v.y);
    atomicAdd(a + 2, v.z);
    atomicAdd(a + 3, v.w);
}

// ------------------------------------------------------- fused SAGE GEMM
// out[16 rows x DOUT] = act( (agg*invdeg) @ Wl + bias + root @ Wr )
// A tiles staged in LDS already in WMMA fragment order (f16); B read from
// pre-swizzled global weights. Inner loop: 1 ds 32B + 1 global 32B + 1 wmma.
template<int DOUT, bool RELU>
__global__ __launch_bounds__((DOUT / 16) * 32)
void sage_gemm_kernel(const float* __restrict__ agg,
                      const float* __restrict__ invdeg,
                      const float* __restrict__ root,
                      const _Float16* __restrict__ WlSw,  // swizzled [DOUT/16][4][32][16]
                      const _Float16* __restrict__ WrSw,  // swizzled
                      const float* __restrict__ bias,     // [DOUT]
                      float* __restrict__ out) {
    constexpr int K  = 128;
    constexpr int NW = DOUT / 16;
    // A fragments, WMMA layout: [buf][kc][lane][i] ; 8 KB total
    __shared__ alignas(32) _Float16 sA[2][K / 32][32][16];

    const int tid  = threadIdx.x;
    const int nthr = NW * 32;
    const int row0 = blockIdx.x * 16;      // N = 100000 = 6250 * 16, no tail

    // Stage both 16x128 A tiles as f16 fragments. Each thread: coalesced
    // float2 global read -> packed 2xf16 -> one ds_store_b32.
    for (int p = tid; p < 2 * 16 * 64; p += nthr) {
        const int buf = p >> 10;           // 0: mean, 1: root
        const int q   = p & 1023;
        const int r   = q >> 6;            // row in tile 0..15
        const int cp  = (q & 63) * 2;      // even column 0..126
        float2 v = *(const float2*)((buf ? root : agg) +
                                    (size_t)(row0 + r) * K + cp);
        if (!buf) {
            const float s = invdeg[row0 + r];
            v.x *= s; v.y *= s;
        }
        const int kc   = cp >> 5;
        const int kk   = cp & 31;
        const int half = kk >> 4;                       // kk[4]
        const int i    = kk & 15;                       // kk[3:0]
        union { _Float16 h[2]; unsigned u; } pk;
        pk.h[0] = (_Float16)v.x;
        pk.h[1] = (_Float16)v.y;
        *(unsigned*)&sA[buf][kc][half * 16 + r][i] = pk.u;
    }
    __syncthreads();

    const int lane = tid & 31;
    const int wv   = tid >> 5;             // wave id = output column tile
    const int l15  = lane & 15;
    const int half = lane >> 4;
    const int col  = wv * 16 + l15;

    // Per-lane base into swizzled weights for this wave's column tile.
    const _Float16* wl = WlSw + ((size_t)wv * (K / 32) * 32 + lane) * 16;
    const _Float16* wr = WrSw + ((size_t)wv * (K / 32) * 32 + lane) * 16;

    v8f c = {};
    #pragma unroll
    for (int kc = 0; kc < K / 32; ++kc) {
        v16h a = *(const v16h*)&sA[0][kc][lane][0];
        v16h b = *(const v16h*)(wl + (size_t)kc * 32 * 16);
        c = __builtin_amdgcn_wmma_f32_16x16x32_f16(false, a, false, b,
                                                   (short)0, c, false, false);
    }
    #pragma unroll
    for (int kc = 0; kc < K / 32; ++kc) {
        v16h a = *(const v16h*)&sA[1][kc][lane][0];
        v16h b = *(const v16h*)(wr + (size_t)kc * 32 * 16);
        c = __builtin_amdgcn_wmma_f32_16x16x32_f16(false, a, false, b,
                                                   (short)0, c, false, false);
    }

    // Epilogue: C VGPR j holds row (j + 8*half), col (lane&15) of the tile.
    const float bv = bias[col];
    #pragma unroll
    for (int j = 0; j < 8; ++j) {
        float v = c[j] + bv;
        if (RELU) v = fmaxf(v, 0.0f);
        out[(size_t)(row0 + j + 8 * half) * DOUT + col] = v;
    }
}

// ---------------------------------------------------------------- launcher
extern "C" void kernel_launch(void* const* d_in, const int* in_sizes, int n_in,
                              void* d_out, int out_size, void* d_ws, size_t ws_size,
                              hipStream_t stream) {
    (void)in_sizes; (void)n_in; (void)out_size; (void)ws_size;

    const float* x    = (const float*)d_in[0];   // [N,128]
    const int*   ei   = (const int*)  d_in[1];   // [2,E] int32 (JAX x64 off)
    const float* W1l  = (const float*)d_in[2];   // [128,128]
    const float* b1   = (const float*)d_in[3];   // [128]
    const float* W1r  = (const float*)d_in[4];   // [128,128]
    const float* W2l  = (const float*)d_in[5];   // [128,64]
    const float* b2   = (const float*)d_in[6];   // [64]
    const float* W2r  = (const float*)d_in[7];   // [128,64]
    const int* srcI = ei;
    const int* dstI = ei + NEDGES;
    float* outF = (float*)d_out;                 // [N,64]

    // Workspace carve-up (256B aligned): ~103 MB total
    char* ws = (char*)d_ws;
    auto carve = [&](size_t bytes) -> char* {
        char* p = ws;
        ws += (bytes + 255) & ~(size_t)255;
        return p;
    };
    float*    agg = (float*)   carve((size_t)NNODES * DFEAT * 4); // reused L1/L2
    float*    h   = (float*)   carve((size_t)NNODES * DFEAT * 4);
    float*    deg = (float*)   carve((size_t)NNODES * 4);
    _Float16* w1l = (_Float16*)carve(128 * 128 * 2);
    _Float16* w1r = (_Float16*)carve(128 * 128 * 2);
    _Float16* w2l = (_Float16*)carve(128 * 64 * 2);
    _Float16* w2r = (_Float16*)carve(128 * 64 * 2);

    const int T = 256;

    // Weights -> f16, pre-swizzled to WMMA B-fragment order (one-time, tiny)
    swizzle_weight_kernel<<<(128 * 128 + T - 1) / T, T, 0, stream>>>(W1l, w1l, 128);
    swizzle_weight_kernel<<<(128 * 128 + T - 1) / T, T, 0, stream>>>(W1r, w1r, 128);
    swizzle_weight_kernel<<<(128 * 64  + T - 1) / T, T, 0, stream>>>(W2l, w2l, 64);
    swizzle_weight_kernel<<<(128 * 64  + T - 1) / T, T, 0, stream>>>(W2r, w2r, 64);

    // Degrees (shared by both layers)
    zero_f32_kernel<<<(NNODES + T - 1) / T, T, 0, stream>>>(deg, NNODES);
    degree_kernel<<<(NEDGES + T - 1) / T, T, 0, stream>>>(dstI, deg, NEDGES);
    invdeg_kernel<<<(NNODES + T - 1) / T, T, 0, stream>>>(deg, NNODES);

    const int nAgg = NNODES * DFEAT;                      // 12.8M
    const int scatterBlocks = (NEDGES * 32 + T - 1) / T;  // 200000

    // ---- Layer 1: h = relu(mean(x) @ W1l + b1 + x @ W1r)
    zero_f32_kernel<<<(nAgg + T - 1) / T, T, 0, stream>>>(agg, nAgg);
    scatter_add_kernel<<<scatterBlocks, T, 0, stream>>>(x, srcI, dstI, agg, NEDGES);
    sage_gemm_kernel<128, true><<<NNODES / 16, 256, 0, stream>>>(
        agg, deg, x, w1l, w1r, b1, h);

    // ---- Layer 2: out = mean(h) @ W2l + b2 + h @ W2r
    zero_f32_kernel<<<(nAgg + T - 1) / T, T, 0, stream>>>(agg, nAgg);
    scatter_add_kernel<<<scatterBlocks, T, 0, stream>>>(h, srcI, dstI, agg, NEDGES);
    sage_gemm_kernel<64, false><<<NNODES / 16, 128, 0, stream>>>(
        agg, deg, h, w2l, w2r, b2, outF);
}
